// AgentEncoder_75840532512965
// MI455X (gfx1250) — matrix operand order repaired and checked by reading, not compiled, not run on hardware
//
#include <hip/hip_runtime.h>

// MI455X / gfx1250: wave32, WMMA f32_16x16x32_f16 path.
// GNN agent-encoder: h=relu(x@Win+b); 2 layers of {ally/enc edge-conv + LN + comb-MLP}.
// Edge MLP decomposed: hidden(e) = relu(Pa[tgt]+Pb[src]+ea@W1c+b1), m = hidden@W2+b2,
// scatter-add to tgt. All 128x128 GEMMs run on v_wmma_f32_16x16x32_f16.
// Weights are pre-swizzled to WMMA-fragment order so every fragment is 2x ds_load_b128.
// Weight tiles staged to LDS via GLOBAL_LOAD_ASYNC_TO_LDS_B128 when available.

typedef __attribute__((ext_vector_type(16))) _Float16 v16h;
typedef __attribute__((ext_vector_type(8)))  _Float16 v8h;
typedef __attribute__((ext_vector_type(8)))  float    v8f;
typedef __attribute__((ext_vector_type(4)))  int      v4i;

#define H    128
#define FIN  39
#define AP   136   // padded LDS A-row stride (halves): breaks 256B same-bank stride

#ifndef __has_builtin
#define __has_builtin(x) 0
#endif
#if __has_builtin(__builtin_amdgcn_global_load_async_to_lds_b128) && \
    __has_builtin(__builtin_amdgcn_s_wait_asynccnt)
#define USE_ASYNC_LDS 1
#else
#define USE_ASYNC_LDS 0
#endif

__device__ __forceinline__ v16h cat8(v8h lo, v8h hi) {
  return __builtin_shufflevector(lo, hi, 0, 1, 2, 3, 4, 5, 6, 7,
                                         8, 9, 10, 11, 12, 13, 14, 15);
}

// Block-wide stage of the pre-swizzled 128x128 f16 weight (32KB linear copy).
__device__ __forceinline__ void stage_B(const _Float16* __restrict__ Wswz,
                                        _Float16* __restrict__ Bs) {
#if USE_ASYNC_LDS
  const v4i* g = (const v4i*)Wswz;
  v4i* l = (v4i*)Bs;
  for (int i = threadIdx.x; i < (H * H) / 8; i += 128) {
    __builtin_amdgcn_global_load_async_to_lds_b128(
        (__attribute__((address_space(1))) v4i*)(unsigned long long)(g + i),
        (__attribute__((address_space(3))) v4i*)(l + i), 0, 0);
  }
  __builtin_amdgcn_s_wait_asynccnt(0);
#else
  const uint4* wsrc = (const uint4*)Wswz;
  uint4* wdst = (uint4*)Bs;
  for (int i = threadIdx.x; i < (H * H) / 8; i += 128) wdst[i] = wsrc[i];
#endif
}

// ---------------------------------------------------------------- weight swizzle
// Output order matches the B-fragment read pattern in wave_gemm_16x128:
//   flat o = ((kk*8 + ct)*32 + lane)*16 + i  holds  W[k][col]
//   with k = kk*32 + ((lane&16)>>1) + (i&7) + ((i&8)<<1), col = ct*16 + (lane&15).
__global__ void cvt_swz_kernel(const float* __restrict__ s, _Float16* __restrict__ d, int n) {
  const int o = blockIdx.x * blockDim.x + threadIdx.x;
  if (o >= n) return;
  const int i    = o & 15;
  const int lane = (o >> 4) & 31;
  const int ct   = (o >> 9) & 7;
  const int kk   = (o >> 12) & 3;
  const int k    = kk * 32 + ((lane & 16) >> 1) + (i & 7) + ((i & 8) << 1);
  const int col  = ct * 16 + (lane & 15);
  d[o] = (_Float16)s[k * H + col];
}

__global__ void copy_f32v4_kernel(float4* __restrict__ d, const float4* __restrict__ s, int n4) {
  int i = blockIdx.x * blockDim.x + threadIdx.x;
  if (i < n4) d[i] = s[i];
}

// ---------------------------------------------------------------- input proj
__global__ void input_proj_kernel(const float* __restrict__ x, const float* __restrict__ Win,
                                  const float* __restrict__ bin, float* __restrict__ h32,
                                  _Float16* __restrict__ h16, int n) {
  __shared__ float xs[FIN];
  const int node = blockIdx.x;
  if (threadIdx.x < FIN) xs[threadIdx.x] = x[(size_t)node * FIN + threadIdx.x];
  __syncthreads();
  const int c = threadIdx.x;
  float acc = bin[c];
  #pragma unroll
  for (int k = 0; k < FIN; ++k) acc = fmaf(xs[k], Win[k * H + c], acc);
  acc = fmaxf(acc, 0.f);
  h32[(size_t)node * H + c] = acc;
  h16[(size_t)node * H + c] = (_Float16)acc;
}

// ---------------------------------------------------------------- wave GEMM
// One wave computes a 16x128 f32 tile: D = A(16 rows x 128 K, LDS row-major padded)
//                                        * B(128x128, LDS in fragment-major swizzle).
// A fragment (ISA 7.12.2): lane row = lane&15; K chunks {hi8+0..7, 16+hi8+0..7} are
// contiguous -> two ds_load_b128. B fragment: 32 contiguous bytes per lane.
// D: element (vgpr v, lane): M = v + 8*(lane>=16), N = lane&15.
__device__ __forceinline__ void wave_gemm_16x128(const _Float16* __restrict__ As,
                                                 const _Float16* __restrict__ Bswz,
                                                 v8f acc[8]) {
  const int lane = threadIdx.x & 31;
  const int hi8  = (lane & 16) >> 1;
  #pragma unroll
  for (int kk = 0; kk < 4; ++kk) {
    const _Float16* Ar = As + (lane & 15) * AP + kk * 32 + hi8;
    const v16h a = cat8(*(const v8h*)Ar, *(const v8h*)(Ar + 16));
    #pragma unroll
    for (int ct = 0; ct < 8; ++ct) {
      const v8h* Bf = (const v8h*)(Bswz + ((size_t)((kk * 8 + ct) * 32 + lane)) * 16);
      const v16h b = cat8(Bf[0], Bf[1]);
      acc[ct] = __builtin_amdgcn_wmma_f32_16x16x32_f16(false, a, false, b, (short)0,
                                                       acc[ct], false, false);
    }
  }
}

// Wave-local stage of 16 A rows (padded stride AP) into LDS.
__device__ __forceinline__ void stage_A(const _Float16* __restrict__ A16,
                                        _Float16* __restrict__ As,
                                        int rowBase, int nrows, int lane) {
  const uint4* asrc = (const uint4*)(A16 + (size_t)rowBase * H);
  uint4* adst = (uint4*)As;  // row stride = AP/8 = 17 uint4
  for (int u = lane; u < 256; u += 32) {
    const int r = u >> 4, c = u & 15;
    uint4 v = make_uint4(0u, 0u, 0u, 0u);
    if (rowBase + r < nrows) v = asrc[u];
    adst[r * (AP / 8) + c] = v;
  }
}

#define NP_CHUNKS   2   // 128 rows per block for node-side GEMMs
#define EDGE_CHUNKS 4   // 256 edges per block for the edge GEMM

// ---------------------------------------------------------------- node proj: P = h16 @ W
__global__ __launch_bounds__(128)
void node_proj_kernel(const _Float16* __restrict__ A16, const _Float16* __restrict__ Wswz,
                      float* __restrict__ out, int nrows) {
  __shared__ __align__(16) _Float16 Bs[H * H];
  __shared__ __align__(16) _Float16 As[4][16 * AP];
  const int lane = threadIdx.x & 31, wave = threadIdx.x >> 5;
  stage_B(Wswz, Bs);
  const int M0 = (lane >> 4) << 3;
  for (int c = 0; c < NP_CHUNKS; ++c) {
    const int rowBase = blockIdx.x * (64 * NP_CHUNKS) + c * 64 + wave * 16;
    stage_A(A16, As[wave], rowBase, nrows, lane);
    __syncthreads();
    v8f acc[8] = {};
    wave_gemm_16x128(As[wave], Bs, acc);
    #pragma unroll
    for (int ct = 0; ct < 8; ++ct)
      #pragma unroll
      for (int v = 0; v < 8; ++v) {
        const int r = rowBase + M0 + v;
        if (r < nrows) out[(size_t)r * H + ct * 16 + (lane & 15)] = acc[ct][v];
      }
  }
}

// ---------------------------------------------------------------- edge conv
// hidden = relu(Pa[tgt]+Pb[src]+ea@W1c+b1); m = hidden@W2+b2; atomic scatter to tgt.
__global__ __launch_bounds__(128)
void edge_conv_kernel(const int* __restrict__ srcIdx, const int* __restrict__ tgtIdx,
                      const float* __restrict__ ea, const float* __restrict__ Pa,
                      const float* __restrict__ Pb, const float* __restrict__ w1c,
                      const float* __restrict__ b1, const _Float16* __restrict__ W2swz,
                      const float* __restrict__ b2, float* __restrict__ accOut, int E_) {
  __shared__ __align__(16) _Float16 Bs[H * H];
  __shared__ __align__(16) _Float16 Hs[4][16 * AP];
  const int lane = threadIdx.x & 31, wave = threadIdx.x >> 5;
  stage_B(W2swz, Bs);
  const int c4 = lane * 4;
  const float4 wc0 = *(const float4*)(w1c + c4);
  const float4 wc1 = *(const float4*)(w1c + H + c4);
  const float4 bb  = *(const float4*)(b1 + c4);
  const int M0 = (lane >> 4) << 3;
  for (int c = 0; c < EDGE_CHUNKS; ++c) {
    const int eBase = blockIdx.x * (64 * EDGE_CHUNKS) + c * 64 + wave * 16;
    for (int el = 0; el < 16; ++el) {
      const int e = eBase + el;
      float4 hv = make_float4(0.f, 0.f, 0.f, 0.f);
      if (e < E_) {
        const int t = tgtIdx[e], s = srcIdx[e];
        const float e0 = ea[2 * e + 0], e1 = ea[2 * e + 1];
        const float4 pa = *(const float4*)(Pa + (size_t)t * H + c4);
        const float4 pb = *(const float4*)(Pb + (size_t)s * H + c4);
        hv.x = fmaxf(pa.x + pb.x + e0 * wc0.x + e1 * wc1.x + bb.x, 0.f);
        hv.y = fmaxf(pa.y + pb.y + e0 * wc0.y + e1 * wc1.y + bb.y, 0.f);
        hv.z = fmaxf(pa.z + pb.z + e0 * wc0.z + e1 * wc1.z + bb.z, 0.f);
        hv.w = fmaxf(pa.w + pb.w + e0 * wc0.w + e1 * wc1.w + bb.w, 0.f);
      }
      _Float16* hr = &Hs[wave][el * AP + c4];
      hr[0] = (_Float16)hv.x; hr[1] = (_Float16)hv.y;
      hr[2] = (_Float16)hv.z; hr[3] = (_Float16)hv.w;
    }
    __syncthreads();
    v8f acc[8] = {};
    wave_gemm_16x128(Hs[wave], Bs, acc);
    #pragma unroll
    for (int ct = 0; ct < 8; ++ct) {
      const int col = ct * 16 + (lane & 15);
      const float bias = b2[col];
      #pragma unroll
      for (int v = 0; v < 8; ++v) {
        const int e = eBase + M0 + v;
        if (e < E_)
          atomicAdd(&accOut[(size_t)tgtIdx[e] * H + col], acc[ct][v] + bias);
      }
    }
  }
}

// ---------------------------------------------------------------- layernorm (wave32 per node)
__global__ void layernorm_kernel(const float* __restrict__ acc, const float* __restrict__ g,
                                 const float* __restrict__ b, float* __restrict__ ln32,
                                 _Float16* __restrict__ ln16, int n) {
  const int lane = threadIdx.x & 31, wave = threadIdx.x >> 5;
  const int node = blockIdx.x * 8 + wave;
  if (node >= n) return;
  const int c4 = lane * 4;
  const float4 v = *(const float4*)(acc + (size_t)node * H + c4);
  float s = v.x + v.y + v.z + v.w;
  float q = v.x * v.x + v.y * v.y + v.z * v.z + v.w * v.w;
  #pragma unroll
  for (int off = 16; off > 0; off >>= 1) {
    s += __shfl_xor(s, off, 32);
    q += __shfl_xor(q, off, 32);
  }
  const float mu  = s * (1.f / H);
  const float var = q * (1.f / H) - mu * mu;
  const float rs  = rsqrtf(var + 1e-5f);
  const float4 gg = *(const float4*)(g + c4);
  const float4 bb = *(const float4*)(b + c4);
  float4 o;
  o.x = (v.x - mu) * rs * gg.x + bb.x;
  o.y = (v.y - mu) * rs * gg.y + bb.y;
  o.z = (v.z - mu) * rs * gg.z + bb.z;
  o.w = (v.w - mu) * rs * gg.w + bb.w;
  *(float4*)(ln32 + (size_t)node * H + c4) = o;
  _Float16* lp = ln16 + (size_t)node * H + c4;
  lp[0] = (_Float16)o.x; lp[1] = (_Float16)o.y;
  lp[2] = (_Float16)o.z; lp[3] = (_Float16)o.w;
}

// ---------------------------------------------------------------- comb: out = ln + relu(ln@W + b)
__global__ __launch_bounds__(128)
void comb_kernel(const _Float16* __restrict__ ln16, const _Float16* __restrict__ Wswz,
                 const float* __restrict__ ln32, const float* __restrict__ cb,
                 float* __restrict__ out32, _Float16* __restrict__ out16, int nrows) {
  __shared__ __align__(16) _Float16 Bs[H * H];
  __shared__ __align__(16) _Float16 As[4][16 * AP];
  const int lane = threadIdx.x & 31, wave = threadIdx.x >> 5;
  stage_B(Wswz, Bs);
  const int M0 = (lane >> 4) << 3;
  for (int c = 0; c < NP_CHUNKS; ++c) {
    const int rowBase = blockIdx.x * (64 * NP_CHUNKS) + c * 64 + wave * 16;
    stage_A(ln16, As[wave], rowBase, nrows, lane);
    __syncthreads();
    v8f acc[8] = {};
    wave_gemm_16x128(As[wave], Bs, acc);
    #pragma unroll
    for (int ct = 0; ct < 8; ++ct) {
      const int col = ct * 16 + (lane & 15);
      const float bias = cb[col];
      #pragma unroll
      for (int v = 0; v < 8; ++v) {
        const int r = rowBase + M0 + v;
        if (r < nrows) {
          const float val = ln32[(size_t)r * H + col] + fmaxf(acc[ct][v] + bias, 0.f);
          out32[(size_t)r * H + col] = val;
          out16[(size_t)r * H + col] = (_Float16)val;
        }
      }
    }
  }
}

// ---------------------------------------------------------------- launch
extern "C" void kernel_launch(void* const* d_in, const int* in_sizes, int n_in,
                              void* d_out, int out_size, void* d_ws, size_t ws_size,
                              hipStream_t stream) {
  (void)n_in; (void)out_size; (void)ws_size;
  const float* x       = (const float*)d_in[0];
  const int*   ally_ei = (const int*)  d_in[1];
  const float* ally_ea = (const float*)d_in[2];
  const int*   enc_ei  = (const int*)  d_in[3];
  const float* enc_ea  = (const float*)d_in[4];
  const float* W_in    = (const float*)d_in[5];
  const float* b_in    = (const float*)d_in[6];
  const float* ally_W1 = (const float*)d_in[7];
  const float* ally_b1 = (const float*)d_in[8];
  const float* ally_W2 = (const float*)d_in[9];
  const float* ally_b2 = (const float*)d_in[10];
  const float* enc_W1  = (const float*)d_in[11];
  const float* enc_b1  = (const float*)d_in[12];
  const float* enc_W2  = (const float*)d_in[13];
  const float* enc_b2  = (const float*)d_in[14];
  const float* ln_g    = (const float*)d_in[15];
  const float* ln_b    = (const float*)d_in[16];
  const float* comb_W  = (const float*)d_in[17];
  const float* comb_b  = (const float*)d_in[18];

  const int N = in_sizes[0] / FIN;
  const int E = in_sizes[1] / 2;
  const size_t NH = (size_t)N * H;

  char* p = (char*)d_ws;
  float*    h32  = (float*)p;      p += NH * 4;
  _Float16* h16  = (_Float16*)p;   p += NH * 2;
  float*    accB = (float*)p;      p += NH * 4;
  float*    ln32 = (float*)p;      p += NH * 4;
  _Float16* ln16 = (_Float16*)p;   p += NH * 2;
  float*    Pa   = (float*)p;      p += NH * 4;
  float*    Pb   = (float*)p;      p += NH * 4;
  _Float16* wts  = (_Float16*)p;

  const int MATN = H * H;  // 16384
  auto w16 = [&](int idx) { return wts + (size_t)idx * MATN; };

  // --- convert + swizzle weight matrices to f16 fragment order
  //     (wts[l*6 + {0:W1a,1:W1b,2:W2}] per type; 12/13 = comb)
  const int cg = (MATN + 255) / 256;
  for (int l = 0; l < 2; ++l) {
    cvt_swz_kernel<<<cg, 256, 0, stream>>>(ally_W1 + (size_t)l * 258 * H,            w16(l * 6 + 0), MATN);
    cvt_swz_kernel<<<cg, 256, 0, stream>>>(ally_W1 + (size_t)l * 258 * H + 128 * H,  w16(l * 6 + 1), MATN);
    cvt_swz_kernel<<<cg, 256, 0, stream>>>(ally_W2 + (size_t)l * MATN,               w16(l * 6 + 2), MATN);
    cvt_swz_kernel<<<cg, 256, 0, stream>>>(enc_W1  + (size_t)l * 258 * H,            w16(l * 6 + 3), MATN);
    cvt_swz_kernel<<<cg, 256, 0, stream>>>(enc_W1  + (size_t)l * 258 * H + 128 * H,  w16(l * 6 + 4), MATN);
    cvt_swz_kernel<<<cg, 256, 0, stream>>>(enc_W2  + (size_t)l * MATN,               w16(l * 6 + 5), MATN);
    cvt_swz_kernel<<<cg, 256, 0, stream>>>(comb_W  + (size_t)l * MATN,               w16(12 + l),    MATN);
  }

  // --- input projection
  input_proj_kernel<<<N, 128, 0, stream>>>(x, W_in, b_in, h32, h16, N);

  const int gr = (N + 64 * NP_CHUNKS - 1) / (64 * NP_CHUNKS);    // node GEMM grid
  const int ge = (E + 64 * EDGE_CHUNKS - 1) / (64 * EDGE_CHUNKS); // edge GEMM grid
  const int gc = (int)(NH / 4 + 255) / 256;                       // copy grid

  for (int l = 0; l < 2; ++l) {
    // acc = h (residual base for conv sums)
    copy_f32v4_kernel<<<gc, 256, 0, stream>>>((float4*)accB, (const float4*)h32, (int)(NH / 4));

    // ally conv
    node_proj_kernel<<<gr, 128, 0, stream>>>(h16, w16(l * 6 + 0), Pa, N);
    node_proj_kernel<<<gr, 128, 0, stream>>>(h16, w16(l * 6 + 1), Pb, N);
    edge_conv_kernel<<<ge, 128, 0, stream>>>(ally_ei, ally_ei + E, ally_ea, Pa, Pb,
        ally_W1 + (size_t)l * 258 * H + 256 * H, ally_b1 + l * H,
        w16(l * 6 + 2), ally_b2 + l * H, accB, E);

    // enc conv
    node_proj_kernel<<<gr, 128, 0, stream>>>(h16, w16(l * 6 + 3), Pa, N);
    node_proj_kernel<<<gr, 128, 0, stream>>>(h16, w16(l * 6 + 4), Pb, N);
    edge_conv_kernel<<<ge, 128, 0, stream>>>(enc_ei, enc_ei + E, enc_ea, Pa, Pb,
        enc_W1 + (size_t)l * 258 * H + 256 * H, enc_b1 + l * H,
        w16(l * 6 + 5), enc_b2 + l * H, accB, E);

    // layernorm
    layernorm_kernel<<<(N + 7) / 8, 256, 0, stream>>>(accB, ln_g + l * H, ln_b + l * H,
                                                      ln32, ln16, N);

    // comb MLP + residual; last layer writes to d_out
    float* outp = (l == 1) ? (float*)d_out : h32;
    comb_kernel<<<gr, 128, 0, stream>>>(ln16, w16(12 + l), ln32, comb_b + l * H,
                                        outp, h16, N);
  }
}
